// EEM_29515015258217
// MI455X (gfx1250) — compile-verified, matrix operand order checked by reading, not run on hardware
//
#include <hip/hip_runtime.h>
#include <hip/hip_bf16.h>
#include <math.h>

typedef __attribute__((ext_vector_type(16))) _Float16 v16h;
typedef __attribute__((ext_vector_type(8)))  _Float16 h8;
typedef __attribute__((ext_vector_type(8)))  float    v8f;

#define IMG 256
#define HW  65536

__device__ __forceinline__ v8f wmma_f16(v16h a, v16h b, v8f c) {
  // D = A(16x32 f16) x B(32x16 f16) + C(16x16 f32)
  return __builtin_amdgcn_wmma_f32_16x16x32_f16(false, a, false, b, (short)0, c,
                                                false, false);
}

// A fragment (16x32 f16). Per lane the 16 elements are two contiguous 16-byte
// runs: rows m0+(lane&15); K sets {0..7,16..23}(+8 for upper half-wave).
// Also serves as the B fragment when the operand is stored [N][Kpad] (B^T),
// since B uses the identical per-lane mapping with N in place of M.
__device__ __forceinline__ v16h frag_a(const _Float16* p, int ld, int m0, int k0,
                                       int lane) {
  const int row = m0 + (lane & 15);
  const int hs  = (lane >> 4) << 3;
  const _Float16* q = p + row * ld + k0 + hs;
  h8 lo = *(const h8*)q;
  h8 hi = *(const h8*)(q + 16);
  return __builtin_shufflevector(lo, hi, 0, 1, 2, 3, 4, 5, 6, 7,
                                 8, 9, 10, 11, 12, 13, 14, 15);
}

// ---- CDNA5 async global->LDS copy (ASYNCcnt path), guarded by __has_builtin
#if __has_builtin(__builtin_amdgcn_global_load_async_to_lds_b128)
#define HAVE_ASYNC_LDS 1
typedef int v4i __attribute__((vector_size(4 * sizeof(int))));
typedef __attribute__((address_space(1))) v4i as1_v4i;  // "__device__" in diags
typedef __attribute__((address_space(3))) v4i as3_v4i;  // "__shared__"  in diags
__device__ __forceinline__ void async_cp16(const void* g, void* l) {
  __builtin_amdgcn_global_load_async_to_lds_b128(
      (as1_v4i*)(unsigned long long)g,
      (as3_v4i*)(unsigned)(unsigned long long)l, 0, 0);
}
__device__ __forceinline__ void wait_async0() {
#if __has_builtin(__builtin_amdgcn_s_wait_asynccnt)
  __builtin_amdgcn_s_wait_asynccnt(0);
#else
  asm volatile("s_wait_asynccnt 0x0" ::: "memory");
#endif
}
#endif

// ---------------------------------------------------------------- k0: zero ws
__global__ void k0_zero(float* p, int n) {
  int i = blockIdx.x * blockDim.x + threadIdx.x;
  if (i < n) p[i] = 0.f;
}

// --------------------------------------------------- k1: window attention+proj
__global__ void k1_window_attn(const float* __restrict__ x,
                               const float* __restrict__ qkv_w,
                               const float* __restrict__ proj_w,
                               const float* __restrict__ proj_b,
                               const float* __restrict__ a2,
                               _Float16* __restrict__ oim) {
  __shared__ __attribute__((aligned(16))) _Float16 xw[64 * 64];  // px x ch(pad)
  __shared__ __attribute__((aligned(16))) _Float16 wqT[48 * 64]; // [N][Kpad]
  __shared__ float qkv[64 * 48];
  __shared__ float olds[64 * 12];

  const int tid = threadIdx.x, lane = tid & 31, wave = tid >> 5;
  const int bid = blockIdx.x;
  const int b = bid >> 10;
  const int g = bid & 1023;
  const int h0 = (g >> 5) * 8, w0 = (g & 31) * 8;

  for (int idx = tid; idx < 64 * 64; idx += 256) {
    int n = idx >> 6, c = idx & 63;
    float v = 0.f;
    if (c < 48) {
      int hh = h0 + (n >> 3), ww = w0 + (n & 7);
      v = x[((size_t)(b * 48 + c) << 16) + hh * IMG + ww];
    }
    xw[idx] = (_Float16)v;
  }
  for (int idx = tid; idx < 48 * 64; idx += 256) {
    int n = idx >> 6, k = idx & 63;
    wqT[idx] = (_Float16)((n < 36 && k < 48) ? qkv_w[n * 48 + k] : 0.f);
  }
  __syncthreads();

  // qkv = xw @ qkv_w^T : M=64, N=36(pad48), K=48(pad64)
  for (int t = wave; t < 12; t += 8) {
    int m0 = (t / 3) * 16, n0 = (t % 3) * 16;
    v8f acc = {};
    for (int k0 = 0; k0 < 64; k0 += 32) {
      v16h a  = frag_a(xw, 64, m0, k0, lane);
      v16h bf = frag_a(wqT, 64, n0, k0, lane);
      acc = wmma_f16(a, bf, acc);
    }
    int n  = n0 + (lane & 15);
    int mb = m0 + ((lane >> 4) << 3);
#pragma unroll
    for (int r = 0; r < 8; ++r) qkv[(mb + r) * 48 + n] = acc[r];
  }
  __syncthreads();

  // softmax attention, dh=3, one thread per (head,row)
  {
    int hh = tid >> 6;
    int n  = tid & 63;
    float q0 = qkv[n * 48 + hh * 3 + 0];
    float q1 = qkv[n * 48 + hh * 3 + 1];
    float q2 = qkv[n * 48 + hh * 3 + 2];
    float sc = a2[hh];
    float p[64];
    float mx = -1e30f;
    for (int m = 0; m < 64; ++m) {
      const float* kr = &qkv[m * 48 + 12 + hh * 3];
      float s = (q0 * kr[0] + q1 * kr[1] + q2 * kr[2]) * sc;
      p[m] = s;
      mx = fmaxf(mx, s);
    }
    float sum = 0.f;
    for (int m = 0; m < 64; ++m) { p[m] = __expf(p[m] - mx); sum += p[m]; }
    float inv = 1.f / sum;
    float o0 = 0.f, o1 = 0.f, o2 = 0.f;
    for (int m = 0; m < 64; ++m) {
      const float* vr = &qkv[m * 48 + 24 + hh * 3];
      float pm = p[m] * inv;
      o0 += pm * vr[0]; o1 += pm * vr[1]; o2 += pm * vr[2];
    }
    olds[n * 12 + hh * 3 + 0] = o0;
    olds[n * 12 + hh * 3 + 1] = o1;
    olds[n * 12 + hh * 3 + 2] = o2;
  }
  __syncthreads();

  // 12->12 proj + bias -> out_im f16 [b][oc][hw]
  for (int idx = tid; idx < 64 * 12; idx += 256) {
    int n = idx / 12, oc = idx % 12;
    float acc = proj_b[oc];
#pragma unroll
    for (int c = 0; c < 12; ++c) acc += olds[n * 12 + c] * proj_w[oc * 12 + c];
    int hh = h0 + (n >> 3), ww = w0 + (n & 7);
    oim[((size_t)(b * 12 + oc) << 16) + hh * IMG + ww] = (_Float16)acc;
  }
}

// ------------------------------------- k2: 1x1 conv (WMMA) + depthwise 3x3
__global__ void k2_imhead(const float* __restrict__ x,
                          const float* __restrict__ w1,   // (96,48)
                          const float* __restrict__ w2,   // (96,1,3,3)
                          _Float16* __restrict__ qv) {    // [b][96][hw]
  __shared__ __attribute__((aligned(16))) _Float16 xh[112 * 64];  // halo px
  __shared__ __attribute__((aligned(16))) _Float16 w1T[96 * 64];  // [N][Kpad]
  __shared__ _Float16 q1h[112 * 96];
  __shared__ float    w2l[96 * 9];

  const int tid = threadIdx.x, lane = tid & 31, wave = tid >> 5;
  const int bid = blockIdx.x;
  const int b = bid >> 10;
  const int t = bid & 1023;
  const int y0 = (t >> 5) * 8, x0 = (t & 31) * 8;

  for (int idx = tid; idx < 112 * 64; idx += 256) {
    int r = idx >> 6, c = idx & 63;
    float v = 0.f;
    if (r < 100 && c < 48) {
      int y = y0 - 1 + r / 10, xx = x0 - 1 + r % 10;
      if (y >= 0 && y < IMG && xx >= 0 && xx < IMG)
        v = x[((size_t)(b * 48 + c) << 16) + y * IMG + xx];
    }
    xh[idx] = (_Float16)v;
  }
  for (int idx = tid; idx < 96 * 64; idx += 256) {
    int o = idx >> 6, k = idx & 63;
    w1T[idx] = (_Float16)((k < 48) ? w1[o * 48 + k] : 0.f);
  }
  for (int idx = tid; idx < 96 * 9; idx += 256) w2l[idx] = w2[idx];
  __syncthreads();

  // q1 = x @ W1^T : M=100(pad112) halo px, N=96, K=48(pad64)
  for (int tt = wave; tt < 42; tt += 8) {
    int m0 = (tt / 6) * 16, n0 = (tt % 6) * 16;
    v8f acc = {};
    for (int k0 = 0; k0 < 64; k0 += 32) {
      v16h a  = frag_a(xh, 64, m0, k0, lane);
      v16h bf = frag_a(w1T, 64, n0, k0, lane);
      acc = wmma_f16(a, bf, acc);
    }
    int n  = n0 + (lane & 15);
    int mb = m0 + ((lane >> 4) << 3);
#pragma unroll
    for (int r = 0; r < 8; ++r) q1h[(mb + r) * 96 + n] = (_Float16)acc[r];
  }
  __syncthreads();

  // depthwise 3x3 on 96 channels over inner 8x8
  for (int idx = tid; idx < 64 * 96; idx += 256) {
    int p = idx / 96, ch = idx % 96;
    int py = p >> 3, px = p & 7;
    float acc = 0.f;
#pragma unroll
    for (int di = 0; di < 3; ++di)
#pragma unroll
      for (int dj = 0; dj < 3; ++dj)
        acc += (float)q1h[((py + di) * 10 + (px + dj)) * 96 + ch] *
               w2l[ch * 9 + di * 3 + dj];
    qv[((size_t)(b * 96 + ch) << 16) + (y0 + py) * IMG + (x0 + px)] =
        (_Float16)acc;
  }
}

// ---------------------- k3: var-norm + 3x3 conv 24->48 as im2col WMMA GEMM
__global__ void k3_ehead(const float* __restrict__ e,
                         const float* __restrict__ gamma,
                         const float* __restrict__ wc,   // (48,24,3,3)
                         _Float16* __restrict__ keg) {   // [b][48][hw]
  __shared__ _Float16 en[100 * 24];
  __shared__ __attribute__((aligned(16))) _Float16 A[64 * 224];   // im2col
  __shared__ __attribute__((aligned(16))) _Float16 BwT[48 * 224]; // [N][Kpad]

  const int tid = threadIdx.x, lane = tid & 31, wave = tid >> 5;
  const int bid = blockIdx.x;
  const int b = bid >> 10;
  const int t = bid & 1023;
  const int y0 = (t >> 5) * 8, x0 = (t & 31) * 8;

  if (tid < 100) {
    int y = y0 - 1 + tid / 10, xx = x0 - 1 + tid % 10;
    if (y >= 0 && y < IMG && xx >= 0 && xx < IMG) {
      float s = 0.f, s2 = 0.f;
      for (int c = 0; c < 24; ++c) {
        float v = e[((size_t)(b * 24 + c) << 16) + y * IMG + xx];
        s += v; s2 += v * v;
      }
      float mean = s * (1.f / 24.f);
      float var  = s2 * (1.f / 24.f) - mean * mean;
      float inv  = rsqrtf(var + 1e-5f);
      for (int c = 0; c < 24; ++c) {
        float v = e[((size_t)(b * 24 + c) << 16) + y * IMG + xx];
        en[tid * 24 + c] = (_Float16)(v * inv * gamma[c]);
      }
    } else {
      for (int c = 0; c < 24; ++c) en[tid * 24 + c] = (_Float16)0.f;
    }
  }
  for (int idx = tid; idx < 48 * 224; idx += 256) {
    int o = idx / 224, kk = idx % 224;
    BwT[idx] = (_Float16)((kk < 216) ? wc[(o * 24 + kk / 9) * 9 + kk % 9] : 0.f);
  }
  __syncthreads();
  for (int idx = tid; idx < 64 * 224; idx += 256) {
    int p = idx / 224, kk = idx % 224;
    _Float16 v = (_Float16)0.f;
    if (kk < 216) {
      int c = kk / 9, rem = kk % 9;
      int hy = (p >> 3) + rem / 3, hx = (p & 7) + rem % 3;
      v = en[(hy * 10 + hx) * 24 + c];
    }
    A[idx] = v;
  }
  __syncthreads();

  // M=64, N=48, K=216(pad224)
  for (int tt = wave; tt < 12; tt += 8) {
    int m0 = (tt / 3) * 16, n0 = (tt % 3) * 16;
    v8f acc = {};
    for (int k0 = 0; k0 < 224; k0 += 32) {
      v16h a  = frag_a(A, 224, m0, k0, lane);
      v16h bf = frag_a(BwT, 224, n0, k0, lane);
      acc = wmma_f16(a, bf, acc);
    }
    int n  = n0 + (lane & 15);
    int mb = m0 + ((lane >> 4) << 3);
#pragma unroll
    for (int r = 0; r < 8; ++r) {
      int p = mb + r;
      int y = y0 + (p >> 3), xx = x0 + (p & 7);
      keg[((size_t)(b * 48 + n) << 16) + y * IMG + xx] = (_Float16)acc[r];
    }
  }
}

// ------ k4: Gram matrices (QK^T, QQ^T, KK^T): async LDS staging + WMMA chains
#define CHUNK 256
__global__ void k4_gram(const _Float16* __restrict__ qv,
                        const _Float16* __restrict__ keg,
                        float* __restrict__ gram) {  // [3][16][256]
  __shared__ __attribute__((aligned(16))) _Float16 qs[16 * CHUNK];
  __shared__ __attribute__((aligned(16))) _Float16 ks[16 * CHUNK];
  const int tid = threadIdx.x, lane = tid & 31, wave = tid >> 5;
  const int bid = blockIdx.x;
  const int b = bid >> 7;
  const int rem = bid & 127;
  const int h = rem >> 5;
  const int s = rem & 31;
  const _Float16* qb = qv  + (size_t)(b * 96 + h * 12) * HW;
  const _Float16* kb = keg + (size_t)(b * 48 + h * 12) * HW;

  // zero padding rows 12..15 once (stay zero across chunks)
  for (int idx = tid; idx < 4 * CHUNK; idx += 256) {
    qs[12 * CHUNK + idx] = (_Float16)0.f;
    ks[12 * CHUNK + idx] = (_Float16)0.f;
  }
  __syncthreads();

  v8f dqk = {}, dqq = {}, dkk = {};
  for (int pix0 = s * 2048; pix0 < s * 2048 + 2048; pix0 += CHUNK) {
#ifdef HAVE_ASYNC_LDS
    // stage 12 q-rows + 12 k-rows: one async b128 per row (32 lanes x 16B)
    for (int slot = wave; slot < 24; slot += 8) {
      int row = (slot < 12) ? slot : slot - 12;
      const _Float16* src =
          ((slot < 12) ? qb : kb) + (size_t)row * HW + pix0 + lane * 8;
      _Float16* dst = ((slot < 12) ? qs : ks) + row * CHUNK + lane * 8;
      async_cp16(src, dst);
    }
    wait_async0();
#else
    for (int idx = tid; idx < 24 * (CHUNK / 8); idx += 256) {
      int slot = idx / (CHUNK / 8), off = (idx % (CHUNK / 8)) * 8;
      int row = (slot < 12) ? slot : slot - 12;
      const _Float16* src = ((slot < 12) ? qb : kb) + (size_t)row * HW + pix0 + off;
      _Float16* dst = ((slot < 12) ? qs : ks) + row * CHUNK + off;
      *(h8*)dst = *(const h8*)src;
    }
#endif
    __syncthreads();
    {
      int k0 = wave * 32;  // each wave owns one 32-pixel K-chunk
      v16h aq = frag_a(qs, CHUNK, 0, k0, lane);
      v16h ak = frag_a(ks, CHUNK, 0, k0, lane);
      dqk = wmma_f16(aq, ak, dqk);
      dqq = wmma_f16(aq, aq, dqq);
      dkk = wmma_f16(ak, ak, dkk);
    }
    __syncthreads();
  }
  int bh = b * 4 + h;
  int n  = lane & 15;
  int mb = (lane >> 4) << 3;
  float* gqk = gram + bh * 256;
  float* gqq = gram + 4096 + bh * 256;
  float* gkk = gram + 8192 + bh * 256;
#pragma unroll
  for (int r = 0; r < 8; ++r) {
    atomicAdd(&gqk[(mb + r) * 16 + n], dqk[r]);
    atomicAdd(&gqq[(mb + r) * 16 + n], dqq[r]);
    atomicAdd(&gkk[(mb + r) * 16 + n], dkk[r]);
  }
}

// ------------------------------------------ k5: normalize + softmax (12x12)
__global__ void k5_attn(const float* __restrict__ gram,
                        const float* __restrict__ a1,
                        float* __restrict__ attn) {
  int tid = threadIdx.x;
  if (tid >= 192) return;
  int b = tid / 48, rem = tid % 48;
  int h = rem / 12, c = rem % 12;
  int bh = b * 4 + h;
  const float* gqk = gram + bh * 256;
  const float* gqq = gram + 4096 + bh * 256;
  const float* gkk = gram + 8192 + bh * 256;
  float nq = fmaxf(sqrtf(fmaxf(gqq[c * 16 + c], 0.f)), 1e-12f);
  float sc = a1[h];
  float row[12];
  float mx = -1e30f;
#pragma unroll
  for (int d = 0; d < 12; ++d) {
    float nk = fmaxf(sqrtf(fmaxf(gkk[d * 16 + d], 0.f)), 1e-12f);
    float v = gqk[c * 16 + d] / (nq * nk) * sc;
    row[d] = v; mx = fmaxf(mx, v);
  }
  float sum = 0.f;
#pragma unroll
  for (int d = 0; d < 12; ++d) { row[d] = __expf(row[d] - mx); sum += row[d]; }
  float inv = 1.f / sum;
#pragma unroll
  for (int d = 0; d < 12; ++d) attn[bh * 144 + c * 12 + d] = row[d] * inv;
}

// ------------- k6: out_eg mix + concat + project_out (WMMA) + residual
__global__ void k6_out(const float* __restrict__ x,
                       const _Float16* __restrict__ oim,
                       const _Float16* __restrict__ qv,
                       const float* __restrict__ attn,
                       const float* __restrict__ pw_g,  // project_out_w (48,60)
                       float* __restrict__ out) {
  __shared__ float attl[576];
  __shared__ __attribute__((aligned(16))) _Float16 cat[128 * 64]; // px x 60(pad)
  __shared__ __attribute__((aligned(16))) _Float16 pwT[48 * 64];  // [N][Kpad]

  const int tid = threadIdx.x, lane = tid & 31, wave = tid >> 5;
  const int bid = blockIdx.x;
  const int b = bid >> 9;
  const int base = (bid & 511) << 7;

  for (int idx = tid; idx < 576; idx += 256) attl[idx] = attn[b * 576 + idx];
  for (int idx = tid; idx < 48 * 64; idx += 256) {
    int o = idx >> 6, c = idx & 63;
    pwT[idx] = (_Float16)((c < 60) ? pw_g[o * 60 + c] : 0.f);
  }
  __syncthreads();

  for (int idx = tid; idx < 128 * 64; idx += 256) {
    int p = idx >> 6, col = idx & 63;
    float v = 0.f;
    if (col < 12) {
      v = (float)oim[((size_t)(b * 12 + col) << 16) + base + p];
    } else if (col < 60) {
      int c2 = col - 12;
      int h = c2 / 12, cc = c2 % 12;
      const float* ar = &attl[h * 144 + cc * 12];
      const _Float16* vb =
          qv + ((size_t)(b * 96 + 48 + h * 12) << 16) + base + p;
#pragma unroll
      for (int d = 0; d < 12; ++d) v += ar[d] * (float)vb[(size_t)d << 16];
    }
    cat[idx] = (_Float16)v;
  }
  __syncthreads();

  // out = x + cat @ Wpo^T : M=128, N=48, K=60(pad64)
  for (int tt = wave; tt < 24; tt += 8) {
    int m0 = (tt / 3) * 16, n0 = (tt % 3) * 16;
    v8f acc = {};
    for (int k0 = 0; k0 < 64; k0 += 32) {
      v16h a  = frag_a(cat, 64, m0, k0, lane);
      v16h bf = frag_a(pwT, 64, n0, k0, lane);
      acc = wmma_f16(a, bf, acc);
    }
    int o  = n0 + (lane & 15);
    int mb = m0 + ((lane >> 4) << 3);
#pragma unroll
    for (int r = 0; r < 8; ++r) {
      size_t gi = ((size_t)(b * 48 + o) << 16) + base + mb + r;
      out[gi] = x[gi] + acc[r];
    }
  }
}

extern "C" void kernel_launch(void* const* d_in, const int* in_sizes, int n_in,
                              void* d_out, int out_size, void* d_ws,
                              size_t ws_size, hipStream_t stream) {
  (void)in_sizes; (void)n_in; (void)out_size; (void)ws_size;
  const float* x     = (const float*)d_in[0];
  const float* e     = (const float*)d_in[1];
  const float* w1    = (const float*)d_in[2];   // imhead_conv1_w (96,48)
  const float* w2    = (const float*)d_in[3];   // imhead_conv2_w (96,1,3,3)
  const float* gamma = (const float*)d_in[4];   // ehead_norm_w (24,)
  const float* wc    = (const float*)d_in[5];   // ehead_conv_w (48,24,3,3)
  const float* a1    = (const float*)d_in[6];
  const float* a2    = (const float*)d_in[7];
  const float* qkvw  = (const float*)d_in[8];   // (36,48)
  const float* projw = (const float*)d_in[9];   // (12,12)
  const float* projb = (const float*)d_in[10];  // (12,)
  const float* pwo   = (const float*)d_in[11];  // (48,60)
  float* out = (float*)d_out;

  char* ws = (char*)d_ws;
  size_t off = 0;
  _Float16* qv  = (_Float16*)(ws + off); off += (size_t)4 * 96 * HW * 2;
  _Float16* keg = (_Float16*)(ws + off); off += (size_t)4 * 48 * HW * 2;
  _Float16* oim = (_Float16*)(ws + off); off += (size_t)4 * 12 * HW * 2;
  float* gram = (float*)(ws + off); off += (size_t)3 * 4096 * 4;
  float* attn = (float*)(ws + off); off += (size_t)16 * 144 * 4;

  k0_zero<<<48, 256, 0, stream>>>(gram, 3 * 4096);
  k1_window_attn<<<4096, 256, 0, stream>>>(x, qkvw, projw, projb, a2, oim);
  k2_imhead<<<4096, 256, 0, stream>>>(x, w1, w2, qv);
  k3_ehead<<<4096, 256, 0, stream>>>(e, gamma, wc, keg);
  k4_gram<<<512, 256, 0, stream>>>(qv, keg, gram);
  k5_attn<<<1, 256, 0, stream>>>(gram, a1, attn);
  k6_out<<<2048, 256, 0, stream>>>(x, oim, qv, attn, pwo, out);
}
